// LemInFrameMoE_85744727097786
// MI455X (gfx1250) — compile-verified, matrix operand order checked by reading, not run on hardware
//
#include <hip/hip_runtime.h>
#include <hip/hip_bf16.h>

// ---------------------------------------------------------------------------
// MI455X (gfx1250) fused graph-MoE pipeline, round 4.
// - 64 edges/block, 512 thr = 16 waves: wave = (expert w&7, mpair w>>3).
// - Gathered x-tile (64x384 f16) staged into LDS via
//   global_load_async_to_lds_b128 (ASYNCcnt) — builtin confirmed present;
//   signature per compiler diagnostic: (v4i AS1*, v4i AS3*, imm, imm).
// - Stage-1 A-fragments from LDS (ds_load_b128); W_moe^T streamed from L2.
// ---------------------------------------------------------------------------

typedef __attribute__((ext_vector_type(16))) _Float16 v16h;
typedef __attribute__((ext_vector_type(8)))  _Float16 v8h;
typedef __attribute__((ext_vector_type(8)))  float    v8f;
typedef int v4i_g __attribute__((vector_size(16)));   // matches builtin proto

#define NNODES 32768
#define NEDGES 131072
#define DF     128
#define KOH    64
#define NEXPT  8
#define HIDD   128
#define K3D    384            // 3*DF
#define XST    392            // x-tile LDS row stride in halfs (784B, +4 banks/row)

#define AS1 __attribute__((address_space(1)))
#define AS3 __attribute__((address_space(3)))

#if defined(__has_builtin)
#if __has_builtin(__builtin_amdgcn_global_load_async_to_lds_b128)
#define HAVE_ASYNC_LDS 1
#endif
#endif
#ifndef HAVE_ASYNC_LDS
#define HAVE_ASYNC_LDS 0
#endif

__device__ __forceinline__ v16h frag_cat(v8h a0, v8h a1) {
  return __builtin_shufflevector(a0, a1, 0,1,2,3,4,5,6,7,8,9,10,11,12,13,14,15);
}

// 16B global -> LDS copy chunk; async when available.
__device__ __forceinline__ void copy16_to_lds(const _Float16* src, _Float16* dst) {
#if HAVE_ASYNC_LDS
  __builtin_amdgcn_global_load_async_to_lds_b128(
      (AS1 v4i_g*)(unsigned long long)(const void*)src,
      (AS3 v4i_g*)dst,
      0, 0);
#else
  *(v8h*)dst = *(const v8h*)src;
#endif
}

__device__ __forceinline__ void wait_async_lds() {
#if HAVE_ASYNC_LDS
#if __has_builtin(__builtin_amdgcn_s_wait_asynccnt)
  __builtin_amdgcn_s_wait_asynccnt(0);
#else
  asm volatile("s_wait_asynccnt 0x0" ::: "memory");
#endif
#endif
}

// ---------------------------------------------------------------------------
// Separable LayerNorm -> f16. One wave per 128-elem row (wave32 shuffles).
// ---------------------------------------------------------------------------
__global__ __launch_bounds__(256) void ln_f16_kernel(
    const float* __restrict__ x, const float* __restrict__ g,
    const float* __restrict__ b, _Float16* __restrict__ out, int rows)
{
  const int row = blockIdx.x * (blockDim.x >> 5) + (threadIdx.x >> 5);
  if (row >= rows) return;
  const int l = threadIdx.x & 31;
  const float* p = x + (size_t)row * DF;
  float v[4];
  float s = 0.0f;
  #pragma unroll
  for (int i = 0; i < 4; ++i) { v[i] = p[l + 32*i]; s += v[i]; }
  #pragma unroll
  for (int off = 16; off > 0; off >>= 1) s += __shfl_xor(s, off, 32);
  const float mu = s * (1.0f / DF);
  float ss = 0.0f;
  #pragma unroll
  for (int i = 0; i < 4; ++i) { v[i] -= mu; ss += v[i] * v[i]; }
  #pragma unroll
  for (int off = 16; off > 0; off >>= 1) ss += __shfl_xor(ss, off, 32);
  const float inv = rsqrtf(ss * (1.0f / DF) + 1e-8f);
  #pragma unroll
  for (int i = 0; i < 4; ++i) {
    const int k = l + 32*i;
    out[(size_t)row * DF + k] = (_Float16)(v[i] * inv * g[k] + b[k]);
  }
}

// ---------------------------------------------------------------------------
// Router: gate = softmax(lat @ W_router); also emit f16 latents.
// ---------------------------------------------------------------------------
__global__ __launch_bounds__(256) void gate_kernel(
    const float* __restrict__ lat, const float* __restrict__ Wr,
    float* __restrict__ gate, _Float16* __restrict__ lat_h)
{
  const int e = blockIdx.x * (blockDim.x >> 5) + (threadIdx.x >> 5);
  if (e >= NEDGES) return;
  const int l = threadIdx.x & 31;
  const float* p = lat + (size_t)e * DF;
  float acc[NEXPT];
  #pragma unroll
  for (int j = 0; j < NEXPT; ++j) acc[j] = 0.0f;
  #pragma unroll
  for (int i = 0; i < 4; ++i) {
    const int k = l + 32*i;
    const float lv = p[k];
    lat_h[(size_t)e * DF + k] = (_Float16)lv;
    #pragma unroll
    for (int j = 0; j < NEXPT; ++j) acc[j] += lv * Wr[k * NEXPT + j];
  }
  #pragma unroll
  for (int j = 0; j < NEXPT; ++j) {
    #pragma unroll
    for (int off = 16; off > 0; off >>= 1) acc[j] += __shfl_xor(acc[j], off, 32);
  }
  float mx = acc[0];
  #pragma unroll
  for (int j = 1; j < NEXPT; ++j) mx = fmaxf(mx, acc[j]);
  float ex[NEXPT];
  float sum = 0.0f;
  #pragma unroll
  for (int j = 0; j < NEXPT; ++j) { ex[j] = __expf(acc[j] - mx); sum += ex[j]; }
  const float rs = 1.0f / sum;
  if (l == 0) {
    #pragma unroll
    for (int j = 0; j < NEXPT; ++j) gate[(size_t)e * NEXPT + j] = ex[j] * rs;
  }
}

// ---------------------------------------------------------------------------
// Weight preps: transpose + f32->f16.
// ---------------------------------------------------------------------------
__global__ __launch_bounds__(256) void wmoeT_kernel(const float* __restrict__ W,
                                                    _Float16* __restrict__ out)
{
  const int idx = blockIdx.x * blockDim.x + threadIdx.x;   // 1024*384
  if (idx >= 1024 * K3D) return;
  const int Nrow = idx / K3D, kk = idx % K3D;
  const int n = Nrow >> 7, h = Nrow & 127;
  out[idx] = (_Float16)W[((size_t)n * K3D + kk) * HIDD + h];
}

__global__ __launch_bounds__(256) void t128_kernel(const float* __restrict__ W,
                                                   _Float16* __restrict__ out)
{
  const int idx = blockIdx.x * blockDim.x + threadIdx.x;   // 128*128
  if (idx >= DF * DF) return;
  const int j = idx >> 7, k = idx & 127;
  out[idx] = (_Float16)W[k * DF + j];
}

// ---------------------------------------------------------------------------
// Fused kernel: 64 edges/block, 512 threads = 16 waves.
//   Stage 0: async-gather x-tile [64 x 384] f16 into LDS  (ASYNCcnt path)
//   Stage 1: z = x @ W_moe^T   (A from LDS; 2 M-tiles x 8 N-tiles x 12 K)
//   Stage 2: gate-weighted cross-expert LDS reduction (+b_moe), silu -> f16
//   Stage 3: edge_messages = silu(msg) @ W_post + b_post  (WMMA, A from LDS)
//   Stage 4: env = lat @ W_env + b_env                    (WMMA)
//   Stage 5: store edge_messages; atomic scatter em*env into node accumulator
// ---------------------------------------------------------------------------
__global__ __launch_bounds__(512) void moe_fused_kernel(
    const _Float16* __restrict__ nn_h, const _Float16* __restrict__ ne_h,
    const _Float16* __restrict__ lat_h, const float* __restrict__ gate,
    const _Float16* __restrict__ WmoeT, const float* __restrict__ b_moe,
    const _Float16* __restrict__ WpostT, const float* __restrict__ b_post,
    const _Float16* __restrict__ WenvT, const float* __restrict__ b_env,
    const int* __restrict__ ec, const int* __restrict__ en,
    float* __restrict__ out_edge, float* __restrict__ agg)
{
  __shared__ alignas(16) _Float16 xtile[64 * XST];      // 49 KB gathered x-tile
  __shared__ alignas(16) float    lds_msg[64 * 132];    // 33.8 KB, padded
  // msgh overlays xtile: x is dead after stage 1; msgh written after barrier
  _Float16* lds_msgh = xtile;

  const int tid   = threadIdx.x;
  const int wave  = tid >> 5;        // 0..15
  const int wexp  = wave & 7;        // expert id / N-tile group
  const int mpair = wave >> 3;       // 0..1 -> M-tiles {2*mpair, 2*mpair+1}
  const int l     = tid & 31;
  const int l15   = l & 15;
  const int lhi   = l >> 4;
  const int klo   = lhi * 8;         // ISA 16-bit frag K offset per lane half
  const int e0    = blockIdx.x * 64;

  // ---- Stage 0: async gather x = [nn[ec] | ne[e] | nn[en]] into LDS.
  // 64 rows x 768B = 3072 x 16B chunks; 6 per thread.
  #pragma unroll
  for (int i = 0; i < 6; ++i) {
    const int c      = tid + i * 512;
    const int row    = c / 48;          // 48 chunks per 768B row
    const int within = c % 48;
    const int seg    = within >> 4;     // 0: nn[ec], 1: ne[e], 2: nn[en]
    const int so     = (within & 15) * 8;   // half offset within segment
    const int erow   = e0 + row;
    const int srow   = (seg == 0) ? ec[erow] : ((seg == 1) ? erow : en[erow]);
    const _Float16* sbase = (seg == 1) ? ne_h : nn_h;
    copy16_to_lds(sbase + (size_t)srow * DF + so, &xtile[row * XST + within * 8]);
  }
  for (int i = tid; i < 64 * 132; i += 512) lds_msg[i] = 0.0f;

  // Pre-warm stage-3/4 weight rows (global_prefetch_b8)
  const _Float16* b2row = WpostT + (size_t)(wexp * 16 + l15) * DF + klo;
  const _Float16* b3row = WenvT  + (size_t)(wexp * 16 + l15) * DF + klo;
  __builtin_prefetch((const void*)b2row, 0, 3);
  __builtin_prefetch((const void*)b3row, 0, 3);

  wait_async_lds();      // drain this wave's async copies
  __syncthreads();       // make all waves' chunks visible

  const v8f vz = {0.f,0.f,0.f,0.f,0.f,0.f,0.f,0.f};
  v8f acc0[8], acc1[8];
  #pragma unroll
  for (int t = 0; t < 8; ++t) { acc0[t] = vz; acc1[t] = vz; }

  // ---- Stage 1: z = x @ W_moe^T, K = 384 (12 blocks of 32), A from LDS
  const _Float16* btw = WmoeT + (size_t)(wexp * 128) * K3D;
  const _Float16* arow0 = &xtile[(mpair * 32 + l15) * XST + klo];
  const _Float16* arow1 = arow0 + 16 * XST;
  #pragma unroll
  for (int kb = 0; kb < 12; ++kb) {
    const v16h a0 = frag_cat(*(const v8h*)(arow0 + kb * 32),
                             *(const v8h*)(arow0 + kb * 32 + 16));
    const v16h a1 = frag_cat(*(const v8h*)(arow1 + kb * 32),
                             *(const v8h*)(arow1 + kb * 32 + 16));
    const _Float16* bcol = btw + (size_t)l15 * K3D + (kb * 32 + klo);
    v16h b = frag_cat(*(const v8h*)bcol, *(const v8h*)(bcol + 16));
    #pragma unroll
    for (int t = 0; t < 8; ++t) {
      v16h bnext = b;
      if (t < 7) {   // software pipeline: issue t+1's load before t's WMMAs
        const _Float16* bp = bcol + (size_t)(t + 1) * 16 * K3D;
        bnext = frag_cat(*(const v8h*)bp, *(const v8h*)(bp + 16));
      }
      acc0[t] = __builtin_amdgcn_wmma_f32_16x16x32_f16(
          false, a0, false, b, (short)0, acc0[t], false, false);
      acc1[t] = __builtin_amdgcn_wmma_f32_16x16x32_f16(
          false, a1, false, b, (short)0, acc1[t], false, false);
      b = bnext;
    }
  }

  // ---- Stage 2: gate-weighted cross-expert reduction into LDS (+ b_moe)
  float gate0[8], gate1[8], bmv[8];
  #pragma unroll
  for (int r = 0; r < 8; ++r) {
    gate0[r] = gate[(size_t)(e0 + mpair * 32 + lhi * 8 + r) * NEXPT + wexp];
    gate1[r] = gate[(size_t)(e0 + mpair * 32 + 16 + lhi * 8 + r) * NEXPT + wexp];
  }
  #pragma unroll
  for (int t = 0; t < 8; ++t)
    bmv[t] = b_moe[wexp * HIDD + t * 16 + l15];

  #pragma unroll
  for (int t = 0; t < 8; ++t) {
    const int c = t * 16 + l15;
    #pragma unroll
    for (int r = 0; r < 8; ++r) {
      const int m0 = mpair * 32 + lhi * 8 + r;        // C-frag M = r + 8*lhi
      atomicAdd(&lds_msg[m0 * 132 + c],        gate0[r] * (acc0[t][r] + bmv[t]));
      atomicAdd(&lds_msg[(m0 + 16) * 132 + c], gate1[r] * (acc1[t][r] + bmv[t]));
    }
  }
  __syncthreads();   // after this, xtile is dead -> safe to write msgh overlay

  // silu + f16 convert for stage 3
  for (int i = tid; i < 64 * 128; i += 512) {
    const int m = i >> 7, c = i & 127;
    const float v = lds_msg[m * 132 + c];
    lds_msgh[m * 136 + c] = (_Float16)(v / (1.0f + __expf(-v)));
  }
  __syncthreads();

  // ---- Stages 3+4: W_post / W_env GEMMs (K=128), wave N-tile = wexp*16
  v8f acc2[2] = {vz, vz}, acc3[2] = {vz, vz};
  #pragma unroll
  for (int kb = 0; kb < 4; ++kb) {
    const v16h b2 = frag_cat(*(const v8h*)(b2row + kb * 32),
                             *(const v8h*)(b2row + kb * 32 + 16));
    const v16h b3 = frag_cat(*(const v8h*)(b3row + kb * 32),
                             *(const v8h*)(b3row + kb * 32 + 16));
    #pragma unroll
    for (int mt = 0; mt < 2; ++mt) {
      const int mrow = mpair * 32 + mt * 16 + l15;
      const _Float16* a2p = lds_msgh + mrow * 136 + klo + kb * 32;
      const v16h a2 = frag_cat(*(const v8h*)a2p, *(const v8h*)(a2p + 16));
      acc2[mt] = __builtin_amdgcn_wmma_f32_16x16x32_f16(
          false, a2, false, b2, (short)0, acc2[mt], false, false);
      const _Float16* a3p = lat_h + (size_t)(e0 + mrow) * DF + klo + kb * 32;
      const v16h a3 = frag_cat(*(const v8h*)a3p, *(const v8h*)(a3p + 16));
      acc3[mt] = __builtin_amdgcn_wmma_f32_16x16x32_f16(
          false, a3, false, b3, (short)0, acc3[mt], false, false);
    }
  }

  // ---- Stage 5: emit edge messages + env-weighted scatter to center nodes
  const int h = wexp * 16 + l15;
  const float bpv = b_post[h];
  const float bev = b_env[h];
  #pragma unroll
  for (int mt = 0; mt < 2; ++mt) {
    #pragma unroll
    for (int r = 0; r < 8; ++r) {
      const int m = mpair * 32 + mt * 16 + r + lhi * 8;
      const int e = e0 + m;
      const float em  = acc2[mt][r] + bpv;   // edge_messages (pre-env)
      const float env = acc3[mt][r] + bev;
      out_edge[(size_t)e * DF + h] = em;
      atomicAdd(&agg[(size_t)ec[e] * DF + h], em * env);
    }
  }
}

// ---------------------------------------------------------------------------
// Final node update: residual mix + onehot elementwise tensor product.
// ---------------------------------------------------------------------------
__global__ __launch_bounds__(128) void node_final_kernel(
    const float* __restrict__ nf, const float* __restrict__ agg,
    const float* __restrict__ onehot, const float* __restrict__ w_onehot,
    float* __restrict__ out_node)
{
  __shared__ float oh[KOH];
  const int v = blockIdx.x;
  const int hh = threadIdx.x;    // 0..127
  if (hh < KOH) oh[hh] = onehot[(size_t)v * KOH + hh];
  __syncthreads();
  float s = 0.0f;
  const float* wrow = w_onehot + (size_t)hh * KOH;
  #pragma unroll 8
  for (int k = 0; k < KOH; ++k) s += oh[k] * wrow[k];
  const float c_old = 0.89442719099991588f;       // rsqrt(1.25), u = 0.5
  const float c_new_half = 0.22360679774897897f;  // c_new * (1/sqrt(AVG_NEIGH))
  const size_t idx = (size_t)v * DF + hh;
  const float val = c_old * nf[idx] + c_new_half * agg[idx];
  out_node[idx] = val * (1.0f + s);
}

// ---------------------------------------------------------------------------
extern "C" void kernel_launch(void* const* d_in, const int* in_sizes, int n_in,
                              void* d_out, int out_size, void* d_ws, size_t ws_size,
                              hipStream_t stream) {
  const float* latents       = (const float*)d_in[0];
  const float* node_features = (const float*)d_in[1];
  const float* edge_features = (const float*)d_in[2];
  const float* node_onehot   = (const float*)d_in[3];
  const int*   edge_index    = (const int*)d_in[6];   // [2, E]
  const float* gamma_n       = (const float*)d_in[8];
  const float* beta_n        = (const float*)d_in[9];
  const float* gamma_e       = (const float*)d_in[10];
  const float* beta_e        = (const float*)d_in[11];
  const float* W_router      = (const float*)d_in[12];
  const float* W_moe         = (const float*)d_in[13];
  const float* b_moe         = (const float*)d_in[14];
  const float* W_post        = (const float*)d_in[15];
  const float* b_post        = (const float*)d_in[16];
  const float* W_env         = (const float*)d_in[17];
  const float* b_env         = (const float*)d_in[18];
  const float* w_onehot      = (const float*)d_in[19];

  const int* ec = edge_index;            // centers
  const int* en = edge_index + NEDGES;   // neighbors

  char* ws = (char*)d_ws;
  _Float16* nn_h   = (_Float16*)ws; ws += (size_t)NNODES * DF * 2;
  _Float16* ne_h   = (_Float16*)ws; ws += (size_t)NEDGES * DF * 2;
  _Float16* lat_h  = (_Float16*)ws; ws += (size_t)NEDGES * DF * 2;
  float*    gate   = (float*)ws;    ws += (size_t)NEDGES * NEXPT * 4;
  _Float16* WmoeT  = (_Float16*)ws; ws += (size_t)1024 * K3D * 2;
  _Float16* WpostT = (_Float16*)ws; ws += (size_t)DF * DF * 2;
  _Float16* WenvT  = (_Float16*)ws; ws += (size_t)DF * DF * 2;
  float*    agg    = (float*)ws;    ws += (size_t)NNODES * DF * 4;

  float* out_node = (float*)d_out;
  float* out_edge = out_node + (size_t)NNODES * DF;

  (void)hipMemsetAsync(agg, 0, (size_t)NNODES * DF * sizeof(float), stream);

  ln_f16_kernel<<<NNODES / 8, 256, 0, stream>>>(node_features, gamma_n, beta_n, nn_h, NNODES);
  ln_f16_kernel<<<NEDGES / 8, 256, 0, stream>>>(edge_features, gamma_e, beta_e, ne_h, NEDGES);

  gate_kernel<<<NEDGES / 8, 256, 0, stream>>>(latents, W_router, gate, lat_h);

  wmoeT_kernel<<<(1024 * K3D + 255) / 256, 256, 0, stream>>>(W_moe, WmoeT);
  t128_kernel<<<(DF * DF + 255) / 256, 256, 0, stream>>>(W_post, WpostT);
  t128_kernel<<<(DF * DF + 255) / 256, 256, 0, stream>>>(W_env, WenvT);

  moe_fused_kernel<<<NEDGES / 64, 512, 0, stream>>>(
      nn_h, ne_h, lat_h, gate, WmoeT, b_moe, WpostT, b_post, WenvT, b_env,
      ec, en, out_edge, agg);

  node_final_kernel<<<NNODES, 128, 0, stream>>>(
      node_features, agg, node_onehot, w_onehot, out_node);
}
